// DGLJTNNEncoder_6124623364173
// MI455X (gfx1250) — matrix operand order repaired and checked by reading, not compiled, not run on hardware
//
#include <hip/hip_runtime.h>
#include <hip/hip_bf16.h>
#include <math.h>

typedef _Float16 f16;
typedef __attribute__((ext_vector_type(16))) _Float16 v16h;
typedef __attribute__((ext_vector_type(8)))  _Float16 v8h;
typedef __attribute__((ext_vector_type(8)))  float    v8f;

#define H   256
#define NB  512          // number of trees (B)
#define NT  127          // nodes per tree
#define EUP 126          // upward edges per tree
#define PADROW (NB*EUP)  // 64512 : zero pad row index in compact up-edge state

#define WMMA_F16(a,b,c) __builtin_amdgcn_wmma_f32_16x16x32_f16(false,(a),false,(b),(short)0,(c),false,false)

__device__ __forceinline__ float sigmoidf_(float x){ return 1.0f/(1.0f + __expf(-x)); }

// A-operand (16x32 f16) from an LDS tile (row stride H), contiguous k order (same as load_b_wt):
// lanes 0-15: row=lane, K = k0..k0+15
// lanes 16-31: row=lane-16, K = k0+16..k0+31
__device__ __forceinline__ v16h load_a_lds(const f16* base, int k0, int lane){
  int row = lane & 15;
  int kb  = k0 + ((lane >> 4) << 4);
  const f16* p = base + row * H + kb;
  v8h lo = *(const v8h*)(p);        // 16B aligned
  v8h hi = *(const v8h*)(p + 8);    // 16B aligned
  v16h a;
#pragma unroll
  for (int j = 0; j < 8; ++j){ a[j] = lo[j]; a[8 + j] = hi[j]; }
  return a;
}

// B-operand (32x16 f16) from transposed weight WT[N][Kld] (row = output col):
// lanes 0-15: col=lane, K = k0..k0+15 ; lanes 16-31: col=lane-16, K = k0+16..k0+31
__device__ __forceinline__ v16h load_b_wt(const f16* wt, int ldk, int n0, int k0, int lane){
  int n  = n0 + (lane & 15);
  int kb = k0 + ((lane >> 4) << 4);
  return *(const v16h*)(wt + (size_t)n * ldk + kb);   // 32B contiguous per lane
}

// ---- weight pre-transpose fp32[K][N] -> f16[N][K] (done once per launch) ----
__global__ void transpose_to_f16(const float* __restrict__ src, f16* __restrict__ dst,
                                 int K, int N){
  int idx = blockIdx.x * blockDim.x + threadIdx.x;
  if (idx >= K * N) return;
  int k = idx / N, n = idx - k * N;            // coalesced read
  dst[(size_t)n * K + k] = (f16)src[idx];
}

__global__ void zero_pad_row(f16* mg, f16* rmg){
  int t = threadIdx.x;
  mg [(size_t)PADROW * H + t] = (f16)0.f;
  rmg[(size_t)PADROW * H + t] = (f16)0.f;
}

// ---- one upward BFS level: 32 edges per workgroup (2 M-tiles), 8 waves ----
// Each wave owns 2 N-tiles of 16 cols; each weight B-fragment feeds 2 WMMAs.
__global__ __launch_bounds__(256)
void level_kernel(const int* __restrict__ wid, const float* __restrict__ emb,
                  const f16* __restrict__ wzT, const f16* __restrict__ whT,
                  const f16* __restrict__ wrT, const f16* __restrict__ urT,
                  const float* __restrict__ bz, const float* __restrict__ bh,
                  const float* __restrict__ bur,
                  f16* __restrict__ mg, f16* __restrict__ rmg, int d)
{
  __shared__ alignas(16) f16 SX [32][H];   // src (child) embedding
  __shared__ alignas(16) f16 SS [32][H];   // sum of predecessor messages
  __shared__ alignas(16) f16 AR [32][H];   // pred r*m sum; REUSED as m_new after phase 1
  __shared__ alignas(16) f16 DXs[32][H];   // dst (parent) embedding
  f16* MNb = &AR[0][0];                    // alias: AR is dead once phase-1 GEMMs finish

  const int tid  = threadIdx.x;
  const int lvl  = 1 << d;
  const int cbeg = lvl - 1;

  // ---- phase 0: cooperative staging (each thread: 2 rows x 16 halves) ----
  {
    int seg = tid & 15, off = seg << 4;
    for (int r = tid >> 4; r < 32; r += 16){
      int gidx = blockIdx.x * 32 + r;
      int tree = gidx >> d;
      int ci   = cbeg + (gidx & (lvl - 1));     // child node (heap index)
      int par  = (ci - 1) >> 1;
      const float* sxp = emb + (size_t)wid[tree * NT + ci ] * H + off;
      const float* dxp = emb + (size_t)wid[tree * NT + par] * H + off;
#pragma unroll
      for (int j = 0; j < 16; ++j){ SX[r][off + j] = (f16)sxp[j]; DXs[r][off + j] = (f16)dxp[j]; }

      bool internal = ci < 63;                  // c < 2^D - 1 has child up-edges
      size_t b0 = internal ? (size_t)(tree * EUP + 2 * ci) * H : (size_t)PADROW * H;
      size_t b1 = internal ? b0 + H             : (size_t)PADROW * H;
      v16h m0 = *(const v16h*)(mg  + b0 + off);
      v16h m1 = *(const v16h*)(mg  + b1 + off);
      v16h r0 = *(const v16h*)(rmg + b0 + off);
      v16h r1 = *(const v16h*)(rmg + b1 + off);
#pragma unroll
      for (int j = 0; j < 16; ++j){
        SS[r][off + j] = (f16)((float)m0[j] + (float)m1[j]);
        AR[r][off + j] = (f16)((float)r0[j] + (float)r1[j]);
      }
    }
  }
  __syncthreads();

  const int wave = tid >> 5, lane = tid & 31;
  const int col16 = lane & 15, rb = (lane & 16) ? 8 : 0;

  // ---- phase 1 GEMMs: accZ = [sx|s]Wz, accH = [sx|arm]Wh  (all tiles) ----
  v8f accZ[2][2] = {{{}, {}}, {{}, {}}};   // [n-tile][m-tile]
  v8f accH[2][2] = {{{}, {}}, {{}, {}}};
#pragma unroll
  for (int t = 0; t < 2; ++t){
    int n0 = (wave * 2 + t) * 16;
#pragma unroll
    for (int ks = 0; ks < 8; ++ks){            // K 0..255 : sx half
      v16h a0 = load_a_lds(&SX[0][0],  ks * 32, lane);
      v16h a1 = load_a_lds(&SX[16][0], ks * 32, lane);
      v16h b  = load_b_wt(wzT, 512, n0, ks * 32, lane);
      accZ[t][0] = WMMA_F16(a0, b, accZ[t][0]);
      accZ[t][1] = WMMA_F16(a1, b, accZ[t][1]);
      v16h b2 = load_b_wt(whT, 512, n0, ks * 32, lane);
      accH[t][0] = WMMA_F16(a0, b2, accH[t][0]);
      accH[t][1] = WMMA_F16(a1, b2, accH[t][1]);
    }
#pragma unroll
    for (int ks = 0; ks < 8; ++ks){            // K 256..511 : s / arm half
      v16h s0 = load_a_lds(&SS[0][0],  ks * 32, lane);
      v16h s1 = load_a_lds(&SS[16][0], ks * 32, lane);
      v16h b  = load_b_wt(wzT, 512, n0, 256 + ks * 32, lane);
      accZ[t][0] = WMMA_F16(s0, b, accZ[t][0]);
      accZ[t][1] = WMMA_F16(s1, b, accZ[t][1]);
      v16h g0 = load_a_lds(&AR[0][0],  ks * 32, lane);
      v16h g1 = load_a_lds(&AR[16][0], ks * 32, lane);
      v16h b2 = load_b_wt(whT, 512, n0, 256 + ks * 32, lane);
      accH[t][0] = WMMA_F16(g0, b2, accH[t][0]);
      accH[t][1] = WMMA_F16(g1, b2, accH[t][1]);
    }
  }
  __syncthreads();   // all reads of AR complete before it is overwritten as m_new

  // ---- combine: m_new = (1-z)*s + z*tanh(.) ; write into aliased AR ----
#pragma unroll
  for (int t = 0; t < 2; ++t){
    int col = (wave * 2 + t) * 16 + col16;
    float bzv = bz[col], bhv = bh[col];
#pragma unroll
    for (int mt = 0; mt < 2; ++mt){
#pragma unroll
      for (int i = 0; i < 8; ++i){
        int rr = mt * 16 + rb + i;
        float z  = sigmoidf_(accZ[t][mt][i] + bzv);
        float hc = tanhf(accH[t][mt][i] + bhv);
        float sv = (float)SS[rr][col];
        MNb[rr * H + col] = (f16)(sv + z * (hc - sv));
      }
    }
  }
  __syncthreads();

  // ---- phase 2: r = sig(dx Wr + m_new Ur + bur); write m, r*m ----
#pragma unroll
  for (int t = 0; t < 2; ++t){
    int n0 = (wave * 2 + t) * 16;
    v8f accR[2] = {{}, {}};
#pragma unroll
    for (int ks = 0; ks < 8; ++ks){
      v16h d0 = load_a_lds(&DXs[0][0],  ks * 32, lane);
      v16h d1 = load_a_lds(&DXs[16][0], ks * 32, lane);
      v16h b  = load_b_wt(wrT, 256, n0, ks * 32, lane);
      accR[0] = WMMA_F16(d0, b, accR[0]);
      accR[1] = WMMA_F16(d1, b, accR[1]);
      v16h m0 = load_a_lds(MNb,          ks * 32, lane);
      v16h m1 = load_a_lds(MNb + 16 * H, ks * 32, lane);
      v16h b2 = load_b_wt(urT, 256, n0, ks * 32, lane);
      accR[0] = WMMA_F16(m0, b2, accR[0]);
      accR[1] = WMMA_F16(m1, b2, accR[1]);
    }
    int col = n0 + col16;
    float bv = bur[col];
#pragma unroll
    for (int mt = 0; mt < 2; ++mt){
#pragma unroll
      for (int i = 0; i < 8; ++i){
        int rr   = mt * 16 + rb + i;
        int gidx = blockIdx.x * 32 + rr;
        int tree = gidx >> d;
        int ci   = cbeg + (gidx & (lvl - 1));
        float rv = sigmoidf_(accR[mt][i] + bv);
        float mn = (float)MNb[rr * H + col];
        size_t o = (size_t)(tree * EUP + (ci - 1)) * H + col;
        mg [o] = (f16)mn;
        rmg[o] = (f16)(rv * mn);
      }
    }
  }
}

// ---- roots: h = relu([x_root | m_up1+m_up2] @ Wg + bg), 16 trees/block ----
__global__ __launch_bounds__(256)
void root_kernel(const int* __restrict__ wid, const float* __restrict__ emb,
                 const f16* __restrict__ wgT, const float* __restrict__ bg,
                 const f16* __restrict__ mg, float* __restrict__ out)
{
  __shared__ alignas(16) f16 XR[16][H];
  __shared__ alignas(16) f16 NM[16][H];
  const int tid = threadIdx.x;
  {
    int r = tid >> 4, seg = tid & 15, off = seg << 4;
    int tree = blockIdx.x * 16 + r;
    const float* xp = emb + (size_t)wid[tree * NT] * H + off;
    size_t b0 = (size_t)(tree * EUP) * H;       // compact up(1)=0, up(2)=1
    v16h m0 = *(const v16h*)(mg + b0 + off);
    v16h m1 = *(const v16h*)(mg + b0 + H + off);
#pragma unroll
    for (int j = 0; j < 16; ++j){
      XR[r][off + j] = (f16)xp[j];
      NM[r][off + j] = (f16)((float)m0[j] + (float)m1[j]);
    }
  }
  __syncthreads();
  const int wave = tid >> 5, lane = tid & 31;
  const int col16 = lane & 15, rb = (lane & 16) ? 8 : 0;
  for (int t = 0; t < 2; ++t){
    int n0 = (wave * 2 + t) * 16;
    v8f acc = {};
#pragma unroll
    for (int ks = 0; ks < 8; ++ks){
      acc = WMMA_F16(load_a_lds(&XR[0][0], ks * 32, lane),
                     load_b_wt(wgT, 512, n0, ks * 32, lane), acc);
      acc = WMMA_F16(load_a_lds(&NM[0][0], ks * 32, lane),
                     load_b_wt(wgT, 512, n0, 256 + ks * 32, lane), acc);
    }
    int col = n0 + col16;
    float bv = bg[col];
#pragma unroll
    for (int i = 0; i < 8; ++i){
      int rr = rb + i;
      float v = acc[i] + bv;
      out[(size_t)(blockIdx.x * 16 + rr) * H + col] = v > 0.f ? v : 0.f;
    }
  }
}

extern "C" void kernel_launch(void* const* d_in, const int* in_sizes, int n_in,
                              void* d_out, int out_size, void* d_ws, size_t ws_size,
                              hipStream_t stream)
{
  const int*   wid = (const int*)  d_in[0];
  const float* emb = (const float*)d_in[1];
  const float* Wz  = (const float*)d_in[2];
  const float* bz  = (const float*)d_in[3];
  const float* Wr  = (const float*)d_in[4];
  const float* Ur  = (const float*)d_in[5];
  const float* bur = (const float*)d_in[6];
  const float* Wh  = (const float*)d_in[7];
  const float* bh  = (const float*)d_in[8];
  const float* Wg  = (const float*)d_in[9];
  const float* bg  = (const float*)d_in[10];

  char* ws = (char*)d_ws;
  size_t o = 0;
  f16* wzT = (f16*)(ws + o); o += (size_t)512*256*2;
  f16* whT = (f16*)(ws + o); o += (size_t)512*256*2;
  f16* wrT = (f16*)(ws + o); o += (size_t)256*256*2;
  f16* urT = (f16*)(ws + o); o += (size_t)256*256*2;
  f16* wgT = (f16*)(ws + o); o += (size_t)512*256*2;
  f16* mg  = (f16*)(ws + o); o += (size_t)(PADROW + 1) * H * 2;
  f16* rmg = (f16*)(ws + o);   // + (PADROW+1)*H*2 ; total ~67 MB

  transpose_to_f16<<<512, 256, 0, stream>>>(Wz, wzT, 512, 256);
  transpose_to_f16<<<512, 256, 0, stream>>>(Wh, whT, 512, 256);
  transpose_to_f16<<<256, 256, 0, stream>>>(Wr, wrT, 256, 256);
  transpose_to_f16<<<256, 256, 0, stream>>>(Ur, urT, 256, 256);
  transpose_to_f16<<<512, 256, 0, stream>>>(Wg, wgT, 512, 256);
  zero_pad_row<<<1, 256, 0, stream>>>(mg, rmg);

  for (int d = 6; d >= 1; --d){                 // upward levels, deepest first
    int blocks = (NB << d) / 32;                // 32 edges per workgroup
    level_kernel<<<blocks, 256, 0, stream>>>(wid, emb, wzT, whT, wrT, urT,
                                             bz, bh, bur, mg, rmg, d);
  }
  root_kernel<<<NB / 16, 256, 0, stream>>>(wid, emb, wgT, bg, mg, (float*)d_out);
}
